// CondConv2d_15350213116103
// MI455X (gfx1250) — matrix operand order repaired and checked by reading, not compiled, not run on hardware
//
#include <hip/hip_runtime.h>
#include <math.h>

typedef _Float16 v8h  __attribute__((ext_vector_type(8)));
typedef _Float16 v16h __attribute__((ext_vector_type(16)));
typedef float    v8f  __attribute__((ext_vector_type(8)));

#define BB 32
#define CCH 192
#define HH 64
#define WW 64
#define OO 192
#define EE 8
#define KSTRIDE (OO*CCH*9)   // 331776 weight elements per expert / per sample

// ---------------- kernel 1a: global average pool -> pooled[B*C] -------------
__global__ __launch_bounds__(256)
void pool_kernel(const float* __restrict__ x, float* __restrict__ pooled) {
    __shared__ float red[256];
    const int bc = blockIdx.x;                    // b*C + c
    const float* p = x + (size_t)bc * (HH * WW);
    const int t = threadIdx.x;
    float s = 0.f;
    #pragma unroll
    for (int i = 0; i < 16; ++i) s += p[t + 256 * i];
    red[t] = s;
    __syncthreads();
    for (int off = 128; off > 0; off >>= 1) {
        if (t < off) red[t] += red[t + off];
        __syncthreads();
    }
    if (t == 0) pooled[bc] = red[0] * (1.0f / (HH * WW));
}

// ---------------- kernel 1b: routing = sigmoid(pooled@rw^T + rb) ------------
__global__ __launch_bounds__(256)
void routing_kernel(const float* __restrict__ pooled,
                    const float* __restrict__ rw,
                    const float* __restrict__ rb,
                    float* __restrict__ routing) {
    const int t = threadIdx.x;        // 0..255 -> (b,e)
    const int b = t >> 3, e = t & 7;
    float z = rb[e];
    const float* pv = pooled + b * CCH;
    const float* wv = rw + e * CCH;
    for (int c = 0; c < CCH; ++c) z += pv[c] * wv[c];
    routing[t] = 1.0f / (1.0f + expf(-z));
}

// ------- kernel 2: mix experts -> per-sample f16 weights [b][o][p][c] -------
// unroll 3 taps at a time: 24 live expert values, no spills; (192,4) lifts
// the VGPR cap to 256 so everything stays register-resident
__global__ __launch_bounds__(192, 4)
void mix_kernel(const float* __restrict__ ew,
                const float* __restrict__ routing,
                _Float16* __restrict__ wout) {
    __shared__ float rS[BB * EE];
    const int o = blockIdx.x;    // 0..191
    const int c = threadIdx.x;   // 0..191
    for (int i = threadIdx.x; i < BB * EE; i += blockDim.x) rS[i] = routing[i];
    __syncthreads();
    const float* ewp = ew + (size_t)o * (CCH * 9) + c * 9;
    #pragma unroll 3
    for (int p = 0; p < 9; ++p) {
        float ev[EE];
        #pragma unroll
        for (int e = 0; e < EE; ++e)
            ev[e] = ewp[(size_t)e * KSTRIDE + p];
        for (int b = 0; b < BB; ++b) {
            float s = 0.f;
            #pragma unroll
            for (int e = 0; e < EE; ++e) s += rS[b * EE + e] * ev[e];
            wout[(size_t)b * KSTRIDE + (size_t)(o * 9 + p) * CCH + c] = (_Float16)s;
        }
    }
}

// ---------------- kernel 3: per-sample implicit-GEMM conv via WMMA ----------
// grid: (H=64, O/64=3, B=32), block: 128 threads (4 wave32)
// block tile: 64 O x 64 pixels (one image row); wave tile 32x32 = 2x2 frags
// __launch_bounds__(128, 4): min 4 waves/EU -> up to 256 VGPRs, no spills
__global__ __launch_bounds__(128, 4)
void condconv_wmma(const float* __restrict__ x,
                   const _Float16* __restrict__ wf,
                   float* __restrict__ y) {
    // pixel-major staged tile: [row 0..2][col 0..65][c 0..31], c-stride padded
    // to 40 halfs (80B = 20 dwords -> conflict-free bank walk, 16B aligned)
    __shared__ _Float16 xT[3 * 66 * 40];

    const int h     = blockIdx.x;
    const int oBase = blockIdx.y * 64;
    const int b     = blockIdx.z;
    const int tid   = threadIdx.x;
    const int lane  = tid & 31;
    const int wave  = tid >> 5;
    const int mW    = (wave & 1) * 32;   // wave M offset within 64
    const int nW    = (wave >> 1) * 32;  // wave N offset within 64
    const int l15   = lane & 15;
    const int lhi   = lane >> 4;

    const _Float16* wb = wf + (size_t)b * KSTRIDE;
    const float*    xb = x + (size_t)b * (CCH * HH * WW);

    v8f acc[2][2] = {};

    for (int cc = 0; cc < 6; ++cc) {           // 6 chunks of 32 channels
        __syncthreads();
        // zero the padding columns (image w=-1 and w=64): 3 rows x 32 ch x 2
        for (int idx = tid; idx < 192; idx += 128) {
            int c = idx & 31;
            int q = idx >> 5;                  // 0..5
            int r = q >> 1;
            int col = (q & 1) ? 65 : 0;
            xT[(r * 66 + col) * 40 + c] = (_Float16)0.f;
        }
        // interior: 3 rows x 32 ch x 16 col-groups of float4 = 1536 = 128 x 12
        #pragma unroll 4
        for (int i = 0; i < 12; ++i) {
            int linear = tid + 128 * i;
            int colg = linear & 15;            // fastest -> coalesced 256B runs
            int rc   = linear >> 4;            // 0..95
            int c    = rc & 31;
            int r    = rc >> 5;                // 0..2
            int row  = h + r - 1;
            float4 f = make_float4(0.f, 0.f, 0.f, 0.f);
            if (row >= 0 && row < HH)
                f = *(const float4*)&xb[((size_t)(cc * 32 + c) * HH + row) * WW + colg * 4];
            _Float16* d = &xT[(r * 66 + (colg * 4 + 1)) * 40 + c];
            d[0 * 40] = (_Float16)f.x;
            d[1 * 40] = (_Float16)f.y;
            d[2 * 40] = (_Float16)f.z;
            d[3 * 40] = (_Float16)f.w;
        }
        // warm the cache for the next chunk (global_prefetch_b8)
        if (cc < 5) {
            const float* nf = xb + ((size_t)((cc + 1) * 32 + lane) * HH + h) * WW + wave * 16;
            __builtin_prefetch(nf, 0, 1);
        }
        __syncthreads();

        #pragma unroll
        for (int p = 0; p < 9; ++p) {          // 3x3 taps, K = 32 channels each
            const int kh = p / 3, kw = p % 3;

            // A fragments: f16 weights, ISA 16-bit A 16x32 layout
            v16h afr[2];
            #pragma unroll
            for (int mi = 0; mi < 2; ++mi) {
                int o = oBase + mW + mi * 16 + l15;
                const _Float16* ap = wb + ((size_t)(o * 9 + p) * CCH + cc * 32 + lhi * 8);
                v8h lo = *(const v8h*)ap;        // K = khalf .. khalf+7
                v8h hi = *(const v8h*)(ap + 16); // K = khalf+16 .. khalf+23
                afr[mi] = __builtin_shufflevector(lo, hi,
                    0,1,2,3,4,5,6,7,8,9,10,11,12,13,14,15);
            }
            // B fragments: patches from LDS, ISA 16-bit B 32x16 layout
            v16h bfr[2];
            #pragma unroll
            for (int ni = 0; ni < 2; ++ni) {
                int wpix = nW + ni * 16 + l15;   // lane <-> pixel column N
                const _Float16* bp = &xT[(kh * 66 + (wpix + kw)) * 40 + lhi * 8];
                v8h lo = *(const v8h*)bp;        // K = kb .. kb+7
                v8h hi = *(const v8h*)(bp + 16); // K = kb+16 .. kb+23
                bfr[ni] = __builtin_shufflevector(lo, hi,
                    0,1,2,3,4,5,6,7,8,9,10,11,12,13,14,15);
            }
            #pragma unroll
            for (int mi = 0; mi < 2; ++mi)
                #pragma unroll
                for (int ni = 0; ni < 2; ++ni)
                    acc[mi][ni] = __builtin_amdgcn_wmma_f32_16x16x32_f16(
                        false, afr[mi], false, bfr[ni],
                        (short)0, acc[mi][ni], false, false);
        }
    }

    // epilogue: C/D layout -> VGPR r holds M=r (lanes 0-15) / M=r+8 (16-31)
    float* yb = y + (size_t)b * (OO * HH * WW);
    #pragma unroll
    for (int mi = 0; mi < 2; ++mi)
        #pragma unroll
        for (int ni = 0; ni < 2; ++ni)
            #pragma unroll
            for (int r = 0; r < 8; ++r) {
                int o    = oBase + mW + mi * 16 + (lhi ? r + 8 : r);
                int wcol = nW + ni * 16 + l15;
                yb[((size_t)o * HH + h) * WW + wcol] = acc[mi][ni][r];
            }
}

extern "C" void kernel_launch(void* const* d_in, const int* in_sizes, int n_in,
                              void* d_out, int out_size, void* d_ws, size_t ws_size,
                              hipStream_t stream) {
    (void)in_sizes; (void)n_in; (void)out_size; (void)ws_size;
    const float* x  = (const float*)d_in[0];   // [B,C,H,W]
    const float* ew = (const float*)d_in[1];   // [E, O*C*9]
    const float* rw = (const float*)d_in[2];   // [E, C]
    const float* rb = (const float*)d_in[3];   // [E]
    float* out = (float*)d_out;                // [B,O,H,W]

    float*    pooled  = (float*)d_ws;                      // B*C floats
    float*    routing = pooled + BB * CCH;                 // B*E floats
    _Float16* wfp16   = (_Float16*)((char*)d_ws + 32768);  // B*O*9*C halfs (~21 MB)

    pool_kernel   <<<BB * CCH, 256, 0, stream>>>(x, pooled);
    routing_kernel<<<1,        256, 0, stream>>>(pooled, rw, rb, routing);
    mix_kernel    <<<OO,       CCH, 0, stream>>>(ew, routing, wfp16);

    dim3 g3(HH, OO / 64, BB);
    condconv_wmma<<<g3, 128, 0, stream>>>(x, wfp16, out);
}
